// MultiHeadAttention_79757542687333
// MI455X (gfx1250) — matrix operand order repaired and checked by reading, not compiled, not run on hardware
//
#include <hip/hip_runtime.h>
#include <hip/hip_bf16.h>

// ---------------------------------------------------------------------------
// Causal MHA forward for MI455X (gfx1250, wave32).
// v_wmma_f32_16x16x32_bf16 everywhere; flash-style online softmax. The
// 1/sqrt(d) scale is folded into the Q store, and causal masking only runs in
// the single diagonal k-block, so the steady-state loop is b128 loads + WMMA
// + minimal softmax VALU.
// ---------------------------------------------------------------------------

typedef __attribute__((ext_vector_type(16))) __bf16 v16bf;
typedef __attribute__((ext_vector_type(8)))  __bf16 v8bf;
typedef __attribute__((ext_vector_type(8)))  float  v8f;

#define Bc 2
#define Sc 2048
#define Ec 1024
#define Hc 16
#define Dc 64

static __device__ __forceinline__ v8f wmma_bf16(v16bf a, v16bf b, v8f c) {
  return __builtin_amdgcn_wmma_f32_16x16x32_bf16(
      false, a, false, b, (short)0, c, false, false);
}

// A-fragment (16x32 bf16, contiguous k order): per lane 16 contiguous k.
static __device__ __forceinline__ v16bf load_afrag(const __bf16* p) {
  v8bf lo = *(const v8bf*)(p);
  v8bf hh = *(const v8bf*)(p + 8);
  return __builtin_shufflevector(lo, hh, 0, 1, 2, 3, 4, 5, 6, 7, 8, 9, 10, 11,
                                 12, 13, 14, 15);
}

// ---------------------------------------------------------------------------
// prep: elementwise f32 -> bf16
// ---------------------------------------------------------------------------
__global__ __launch_bounds__(256) void cvt_f32_bf16(
    const float* __restrict__ src, __bf16* __restrict__ dst, int n) {
  const int i = (blockIdx.x * 256 + threadIdx.x) * 8;
  if (i >= n) return;
  const float4 x0 = *(const float4*)(src + i);
  const float4 x1 = *(const float4*)(src + i + 4);
  v8bf o;
  o[0] = (__bf16)x0.x; o[1] = (__bf16)x0.y;
  o[2] = (__bf16)x0.z; o[3] = (__bf16)x0.w;
  o[4] = (__bf16)x1.x; o[5] = (__bf16)x1.y;
  o[6] = (__bf16)x1.z; o[7] = (__bf16)x1.w;
  *(v8bf*)(dst + i) = o;
}

// ---------------------------------------------------------------------------
// prep: W f32 [K,N] -> bf16 [N,K] (LDS 32x32 tile transpose)
// ---------------------------------------------------------------------------
__global__ __launch_bounds__(256) void transpose_to_bf16(
    const float* __restrict__ src, __bf16* __restrict__ dst, int K, int N) {
  __shared__ float tile[32][33];
  const int n0 = blockIdx.x * 32;
  const int k0 = blockIdx.y * 32;
  const int tx = threadIdx.x & 31;
  const int ty = threadIdx.x >> 5;
#pragma unroll
  for (int r = 0; r < 4; ++r)
    tile[ty + r * 8][tx] = src[(size_t)(k0 + ty + r * 8) * N + n0 + tx];
  __syncthreads();
#pragma unroll
  for (int r = 0; r < 4; ++r)
    dst[(size_t)(n0 + ty + r * 8) * K + k0 + tx] =
        (__bf16)tile[tx][ty + r * 8];
}

// ---------------------------------------------------------------------------
// GEMM: C[M,N] = A[M,K] x Bt[N,K]^T + bias, all-bf16 operands, f32 acc.
// MODE 0: Q scaled by 1/sqrt(d) and stored [B,H,S,d]; K stored [B,H,S,d];
//         V stored transposed [B,H,d,S].
// MODE 1: f32 store to fout[M,N].
// ---------------------------------------------------------------------------
template <int MODE>
__global__ __launch_bounds__(256) void gemm_bf16_kernel(
    const __bf16* __restrict__ A, const __bf16* __restrict__ Bt,
    const float* __restrict__ bias,
    __bf16* __restrict__ qo, __bf16* __restrict__ ko, __bf16* __restrict__ vt,
    float* __restrict__ fout, int M, int N, int K) {
  const int lane = threadIdx.x & 31;
  const int wave = threadIdx.x >> 5;
  const int col  = lane & 15;
  const int hi   = lane >> 4;
  const int m0   = blockIdx.x * 16;
  const int n0   = blockIdx.y * 512 + wave * 64;

  v8f acc[4] = {};
  const __bf16* ap = A + (size_t)(m0 + col) * K + hi * 16;
  const __bf16* bp = Bt + (size_t)(n0 + col) * K + hi * 16;

  for (int k0 = 0; k0 < K; k0 += 32) {
    if (k0 + 128 < K) __builtin_prefetch(ap + k0 + 128, 0, 0);
    const v16bf a = load_afrag(ap + k0);
#pragma unroll
    for (int t = 0; t < 4; ++t) {
      const v16bf b = *(const v16bf*)(bp + (size_t)t * 16 * K + k0);
      acc[t] = wmma_bf16(a, b, acc[t]);
    }
  }

  if (MODE == 0) {
    const int which = n0 >> 10;             // 0=Q 1=K 2=V
    const int h     = (n0 >> 6) & (Hc - 1);
    const float sc  = (which == 0) ? 0.125f : 1.0f;  // fold 1/sqrt(64) into Q
#pragma unroll
    for (int t = 0; t < 4; ++t) {
      const int dd   = t * 16 + col;
      const float bv = bias[n0 + t * 16 + col];
#pragma unroll
      for (int j = 0; j < 8; ++j) {
        const int m  = m0 + (hi << 3) + j;
        const int bb = m >> 11;
        const int s  = m & (Sc - 1);
        const __bf16 val = (__bf16)((acc[t][j] + bv) * sc);
        if (which == 2)
          vt[(((size_t)bb * Hc + h) * Dc + dd) * Sc + s] = val;
        else
          ((which == 0) ? qo : ko)[(((size_t)bb * Hc + h) * Sc + s) * Dc + dd] =
              val;
      }
    }
  } else {
#pragma unroll
    for (int t = 0; t < 4; ++t) {
      const int n    = n0 + t * 16 + col;
      const float bv = bias[n];
#pragma unroll
      for (int j = 0; j < 8; ++j) {
        const int m = m0 + (hi << 3) + j;
        fout[(size_t)m * N + n] = acc[t][j] + bv;
      }
    }
  }
}

// ---------------------------------------------------------------------------
// One 64-wide k-block of flash attention for a 16-row q tile.
// MASK=false: steady state (no causal compares at all).
// MASK=true : the single diagonal block.
// 8 WMMA (QK^T) + 8 WMMA (PV) per call.
// ---------------------------------------------------------------------------
template <bool MASK>
static __device__ __forceinline__ void fa_block(
    int kbase, int qbase, int col, int hi, const __bf16* __restrict__ Kp,
    const __bf16* __restrict__ Vt, const v16bf (&qf)[2], v8f (&o)[4],
    float (&mrow)[8], float (&lrow)[8], __bf16* plsw) {
  // ---- S = Q K^T : four 16x16 C tiles over 64 k columns ----
  v8f cc[4] = {};
#pragma unroll
  for (int t = 0; t < 4; ++t) {
    const __bf16* kp = Kp + (size_t)(kbase + t * 16 + col) * Dc + hi * 16;
#pragma unroll
    for (int half = 0; half < 2; ++half) {
      const v16bf kf = *(const v16bf*)(kp + half * 32);  // contiguous B-frag
      cc[t] = wmma_bf16(qf[half], kf, cc[t]);
    }
  }

  // ---- online softmax across 64 columns (once per 16 WMMAs) ----
#pragma unroll
  for (int j = 0; j < 8; ++j) {
    const int rowl = (hi << 3) + j;
    const int row  = qbase + rowl;
    float s[4];
#pragma unroll
    for (int t = 0; t < 4; ++t) {
      s[t] = cc[t][j];
      if (MASK && (kbase + t * 16 + col > row)) s[t] = -1e30f;
    }
    float mx = fmaxf(fmaxf(s[0], s[1]), fmaxf(s[2], s[3]));
#pragma unroll
    for (int msk = 1; msk < 16; msk <<= 1)
      mx = fmaxf(mx, __shfl_xor(mx, msk, 32));  // stays within 16-lane half
    const float mnew = fmaxf(mrow[j], mx);
    const float corr = __expf(mrow[j] - mnew);
    float p[4], rs = 0.0f;
#pragma unroll
    for (int t = 0; t < 4; ++t) { p[t] = __expf(s[t] - mnew); rs += p[t]; }
#pragma unroll
    for (int msk = 1; msk < 16; msk <<= 1) rs += __shfl_xor(rs, msk, 32);
    lrow[j] = lrow[j] * corr + rs;
    mrow[j] = mnew;
#pragma unroll
    for (int t = 0; t < 4; ++t) o[t][j] *= corr;
#pragma unroll
    for (int t = 0; t < 4; ++t)
      plsw[rowl * 64 + t * 16 + col] = (__bf16)p[t];
  }

  // ---- P fragments (two 16x32 A-frags) from LDS ----
  v16bf pf[2];
#pragma unroll
  for (int ph = 0; ph < 2; ++ph)
    pf[ph] = load_afrag(plsw + col * 64 + ph * 32 + hi * 16);

  // ---- O += P V : V^T gives contiguous B-frags ----
#pragma unroll
  for (int t = 0; t < 4; ++t) {
    const __bf16* vp = Vt + (size_t)(t * 16 + col) * Sc + kbase + hi * 16;
#pragma unroll
    for (int ph = 0; ph < 2; ++ph) {
      const v16bf vf = *(const v16bf*)(vp + ph * 32);
      o[t] = wmma_bf16(pf[ph], vf, o[t]);
    }
  }
}

// ---------------------------------------------------------------------------
// Flash attention. One wave per 16-row q tile; k in blocks of 64.
// ---------------------------------------------------------------------------
__global__ __launch_bounds__(128) void flash_attn_kernel(
    const __bf16* __restrict__ Q, const __bf16* __restrict__ Km,
    const __bf16* __restrict__ VT, __bf16* __restrict__ attnb) {
  __shared__ __attribute__((aligned(32))) __bf16 pls[4][16 * 64];

  const int lane  = threadIdx.x & 31;
  const int wave  = threadIdx.x >> 5;
  const int col   = lane & 15;
  const int hi    = lane >> 4;
  const int bh    = blockIdx.x;
  const int b     = bh >> 4;
  const int h     = bh & (Hc - 1);
  const int qbase = (blockIdx.y * 4 + wave) * 16;

  const __bf16* Qp = Q  + (size_t)bh * Sc * Dc;
  const __bf16* Kp = Km + (size_t)bh * Sc * Dc;
  const __bf16* Vt = VT + (size_t)bh * Dc * Sc;
  __bf16* plsw = &pls[wave][0];

  v16bf qf[2];
  {
    const __bf16* qp = Qp + (size_t)(qbase + col) * Dc + hi * 16;
    qf[0] = load_afrag(qp);
    qf[1] = load_afrag(qp + 32);
  }

  v8f o[4] = {};
  float mrow[8], lrow[8];
#pragma unroll
  for (int j = 0; j < 8; ++j) { mrow[j] = -1e30f; lrow[j] = 0.0f; }

  // full (unmasked) 64-blocks, then exactly one masked diagonal block
  const int nfull = qbase >> 6;
  for (int kb = 0; kb < nfull; ++kb)
    fa_block<false>(kb * 64, qbase, col, hi, Kp, Vt, qf, o, mrow, lrow, plsw);
  fa_block<true>(nfull * 64, qbase, col, hi, Kp, Vt, qf, o, mrow, lrow, plsw);

  // ---- finalize: divide by row sums, store bf16 [B,S,E] ----
#pragma unroll
  for (int j = 0; j < 8; ++j) {
    const int row   = qbase + (hi << 3) + j;
    const float inv = 1.0f / lrow[j];
    __bf16* op = attnb + ((size_t)b * Sc + row) * Ec + h * Dc + col;
#pragma unroll
    for (int t = 0; t < 4; ++t) op[t * 16] = (__bf16)(o[t][j] * inv);
  }
}

// ---------------------------------------------------------------------------
extern "C" void kernel_launch(void* const* d_in, const int* in_sizes, int n_in,
                              void* d_out, int out_size, void* d_ws,
                              size_t ws_size, hipStream_t stream) {
  (void)in_sizes; (void)n_in; (void)out_size; (void)ws_size;
  const float* X     = (const float*)d_in[0];
  // d_in[1] = tril mask: causal masking computed analytically, unused.
  const float* W_qkv = (const float*)d_in[2];
  const float* b_qkv = (const float*)d_in[3];
  const float* W_o   = (const float*)d_in[4];
  const float* b_o   = (const float*)d_in[5];
  float* out = (float*)d_out;

  const int M = Bc * Sc;                     // 4096
  const size_t nX   = (size_t)M * Ec;
  const size_t nQKV = (size_t)Bc * Hc * Sc * Dc;

  __bf16* Xbf   = (__bf16*)d_ws;
  __bf16* WqkvT = Xbf + nX;                  // [3E, E]
  __bf16* WoT   = WqkvT + (size_t)3 * Ec * Ec;
  __bf16* Qw    = WoT + (size_t)Ec * Ec;     // [B,H,S,d], pre-scaled
  __bf16* Kw    = Qw + nQKV;                 // [B,H,S,d]
  __bf16* Vt    = Kw + nQKV;                 // [B,H,d,S]
  __bf16* attnb = Vt + nQKV;                 // [B,S,E]

  cvt_f32_bf16<<<(int)(nX / (8 * 256)), 256, 0, stream>>>(X, Xbf, (int)nX);
  transpose_to_bf16<<<dim3(3 * Ec / 32, Ec / 32), 256, 0, stream>>>(
      W_qkv, WqkvT, Ec, 3 * Ec);
  transpose_to_bf16<<<dim3(Ec / 32, Ec / 32), 256, 0, stream>>>(W_o, WoT, Ec,
                                                                Ec);

  gemm_bf16_kernel<0><<<dim3(M / 16, 6), 256, 0, stream>>>(
      Xbf, WqkvT, b_qkv, Qw, Kw, Vt, nullptr, M, 3 * Ec, Ec);

  flash_attn_kernel<<<dim3(Bc * Hc, Sc / 64), 128, 0, stream>>>(Qw, Kw, Vt,
                                                                attnb);

  gemm_bf16_kernel<1><<<dim3(M / 16, 2), 256, 0, stream>>>(
      attnb, WoT, b_o, nullptr, nullptr, nullptr, out, M, Ec, Ec);
}
